// MultiHeadAttention_31439160607075
// MI455X (gfx1250) — matrix operand order repaired and checked by reading, not compiled, not run on hardware
//
#include <hip/hip_runtime.h>
#include <hip/hip_bf16.h>
#include <cstdint>

typedef __attribute__((ext_vector_type(16))) _Float16 v16h;
typedef __attribute__((ext_vector_type(8)))  float    v8f;

static constexpr int NH     = 16;    // heads
static constexpr int HID    = 1024;  // hidden
static constexpr int DH     = 64;    // head dim
static constexpr int NB     = 2;     // batch
static constexpr int NS     = 2048;  // seq
static constexpr int NM     = NB * NS; // 4096 rows for projections

// ---------------------------------------------------------------------------
// f32 -> f16 elementwise convert (vectorized: 4 floats per thread)
__global__ __launch_bounds__(256) void cvt_f16_kernel(const float* __restrict__ src,
                                                      _Float16* __restrict__ dst, int n4) {
    int i = blockIdx.x * blockDim.x + threadIdx.x;
    if (i < n4) {
        float4 v = ((const float4*)src)[i];
        _Float16 h0 = (_Float16)v.x, h1 = (_Float16)v.y;
        _Float16 h2 = (_Float16)v.z, h3 = (_Float16)v.w;
        union { _Float16 h[4]; uint2 u; } pk;
        pk.h[0] = h0; pk.h[1] = h1; pk.h[2] = h2; pk.h[3] = h3;
        ((uint2*)dst)[i] = pk.u;
    }
}

// W (k-major [k][n]) -> WT f16 ([n][k])
__global__ __launch_bounds__(256) void cvt_wt_kernel(const float* __restrict__ w,
                                                     _Float16* __restrict__ wt) {
    int i = blockIdx.x * blockDim.x + threadIdx.x;   // i = n*HID + k
    int n = i >> 10, k = i & (HID - 1);
    wt[i] = (_Float16)w[k * HID + n];
}

// ---------------------------------------------------------------------------
// Projection GEMM: dst = X @ W  (via Xh f16 [m][k], WT f16 [n][k])
// dst layout: transposed==0 -> [h,b,s,d] ; transposed==1 -> [h,b,d,s]
__global__ __launch_bounds__(128) void proj_kernel(const _Float16* __restrict__ Xh,
                                                   const _Float16* __restrict__ WT,
                                                   _Float16* __restrict__ dst,
                                                   int transposed) {
    const int lane = threadIdx.x & 31;
    const int wave = threadIdx.x >> 5;
    const int mt = blockIdx.x * 4 + wave;      // 0..255
    const int nt = blockIdx.y;                 // 0..63
    const int m0 = mt * 16, n0 = nt * 16;
    const int g = lane >> 4, l15 = lane & 15;

    const _Float16* arow = Xh + (size_t)(m0 + l15) * HID;
    const _Float16* brow = WT + (size_t)(n0 + l15) * HID;

    v8f c = {};
    for (int kc = 0; kc < HID; kc += 32) {
        union { uint4 u[2]; v16h h; } a, b;
        a.u[0] = *(const uint4*)(arow + kc + g * 8);
        a.u[1] = *(const uint4*)(arow + kc + 16 + g * 8);
        b.u[0] = *(const uint4*)(brow + kc + g * 8);
        b.u[1] = *(const uint4*)(brow + kc + 16 + g * 8);
        c = __builtin_amdgcn_wmma_f32_16x16x32_f16(false, a.h, false, b.h,
                                                   (short)0, c, false, false);
    }

    const int h  = n0 >> 6;        // head
    const int d0 = n0 & 63;        // head-dim base
    const int bb = m0 / NS;
    const int s0 = m0 % NS;
    if (!transposed) {
        _Float16* base = dst + ((size_t)(h * NB + bb) * NS) * DH;
#pragma unroll
        for (int r = 0; r < 8; ++r) {
            int s = s0 + r + g * 8;
            base[(size_t)s * DH + d0 + l15] = (_Float16)c[r];
        }
    } else {
        _Float16* base = dst + ((size_t)(h * NB + bb) * DH) * NS;
#pragma unroll
        for (int r = 0; r < 8; ++r) {
            int s = s0 + r + g * 8;
            base[(size_t)(d0 + l15) * NS + s] = (_Float16)c[r];
        }
    }
}

// ---------------------------------------------------------------------------
// Fused attention: scores (WMMA) -> causal softmax (LDS) -> aw store -> P@V (WMMA)
__global__ __launch_bounds__(256) void attn_kernel(const _Float16* __restrict__ qbuf, // [h,b,s,d]
                                                   const _Float16* __restrict__ kbuf, // [h,b,s,d]
                                                   const _Float16* __restrict__ vT,   // [h,b,d,s]
                                                   const int* __restrict__ maskp,
                                                   float* __restrict__ aw,            // [h,b,s,s]
                                                   float* __restrict__ otmp) {        // [b,s,hid]
    extern __shared__ float smem[];
    float* sc     = smem;               // 16 * 2048 scores / probs
    float* red    = smem + 16 * NS;     // 16 * 16 partials
    float* rowmax = red + 256;          // 16
    float* rowinv = rowmax + 16;        // 16

    const int qt  = blockIdx.x;         // q tile (16 rows)
    const int hb  = blockIdx.y;         // h*NB + b
    const int tid = threadIdx.x;
    const int lane = tid & 31, wave = tid >> 5;
    const int g = lane >> 4, l15 = lane & 15;
    const int i0 = qt * 16;
    const int msk = *maskp;
    const int imax = i0 + 15;
    const size_t base_s = (size_t)hb * NS;

    // ---- Phase 1: scores = (q k^T) / 8 into LDS -------------------------------
    const _Float16* qrow = qbuf + (base_s + i0 + l15) * DH;
    union { uint4 u[2]; v16h h; } a0, a1;
    a0.u[0] = *(const uint4*)(qrow + 0  + g * 8);
    a0.u[1] = *(const uint4*)(qrow + 16 + g * 8);
    a1.u[0] = *(const uint4*)(qrow + 32 + g * 8);
    a1.u[1] = *(const uint4*)(qrow + 48 + g * 8);

    for (int t = 0; t < 16; ++t) {
        int n0 = wave * 256 + t * 16;
        if (msk && n0 > imax) break;    // fully masked tiles never computed
        const _Float16* krow = kbuf + (base_s + n0 + l15) * DH;
        union { uint4 u[2]; v16h h; } b0, b1;
        b0.u[0] = *(const uint4*)(krow + g * 8);
        b0.u[1] = *(const uint4*)(krow + 16 + g * 8);
        b1.u[0] = *(const uint4*)(krow + 32 + g * 8);
        b1.u[1] = *(const uint4*)(krow + 48 + g * 8);
        v8f c = {};
        c = __builtin_amdgcn_wmma_f32_16x16x32_f16(false, a0.h, false, b0.h,
                                                   (short)0, c, false, false);
        c = __builtin_amdgcn_wmma_f32_16x16x32_f16(false, a1.h, false, b1.h,
                                                   (short)0, c, false, false);
#pragma unroll
        for (int r = 0; r < 8; ++r)
            sc[(r + g * 8) * NS + n0 + l15] = c[r] * 0.125f;
    }
    __syncthreads();

    // ---- Phase 2: causal softmax (unnormalized probs kept in LDS) --------------
    const int r2 = tid >> 4;
    const int s2 = tid & 15;
    const int L = msk ? (i0 + r2 + 1) : NS;   // valid j < L
    float lmax = -3.4e38f;
    for (int cc = 0; cc < 128; ++cc) {
        int j = s2 + (cc << 4);
        if (j < L) lmax = fmaxf(lmax, sc[r2 * NS + j]);
    }
    red[r2 * 16 + s2] = lmax;
    __syncthreads();
    if (s2 == 0) {
        float m = red[r2 * 16];
        for (int k = 1; k < 16; ++k) m = fmaxf(m, red[r2 * 16 + k]);
        rowmax[r2] = m;
    }
    __syncthreads();
    const float mrow = rowmax[r2];
    float lsum = 0.f;
    for (int cc = 0; cc < 128; ++cc) {
        int j = s2 + (cc << 4);
        float p = 0.f;
        if (j < L) { p = __expf(sc[r2 * NS + j] - mrow); lsum += p; }
        sc[r2 * NS + j] = p;
    }
    red[r2 * 16 + s2] = lsum;
    __syncthreads();
    if (s2 == 0) {
        float s = 0.f;
        for (int k = 0; k < 16; ++k) s += red[r2 * 16 + k];
        rowinv[r2] = 1.0f / s;
    }
    __syncthreads();

    // ---- Phase 3: write aw, float4-vectorized (dominant HBM traffic) -----------
    float* awbase = aw + ((size_t)hb * NS + i0) * NS;
    for (int r = 0; r < 16; ++r) {
        const float inv = rowinv[r];
        const float4* srow = (const float4*)(sc + r * NS);
        float4* drow = (float4*)(awbase + (size_t)r * NS);
#pragma unroll
        for (int k = 0; k < 2; ++k) {           // 2048/4 = 512 float4, 256 threads
            int j4 = tid + k * 256;
            float4 v = srow[j4];
            v.x *= inv; v.y *= inv; v.z *= inv; v.w *= inv;
            drow[j4] = v;
        }
    }

    // ---- Phase 4: O = (P @ V) / rowsum  (waves 0..3, 16 cols each) -------------
    if (wave < 4) {
        int kend = msk ? ((i0 + 16 + 31) & ~31) : NS;
        if (kend > NS) kend = NS;
        const _Float16* vbase = vT + ((size_t)hb * DH + wave * 16 + l15) * NS;
        const float* pr = sc + (size_t)l15 * NS;   // A row M = lane&15
        v8f c = {};
        for (int kc = 0; kc < kend; kc += 32) {
            float4 f0 = *(const float4*)(pr + kc + g * 8);
            float4 f1 = *(const float4*)(pr + kc + g * 8 + 4);
            float4 f2 = *(const float4*)(pr + kc + 16 + g * 8);
            float4 f3 = *(const float4*)(pr + kc + 16 + g * 8 + 4);
            v16h ah;
            ah[0]  = (_Float16)f0.x; ah[1]  = (_Float16)f0.y;
            ah[2]  = (_Float16)f0.z; ah[3]  = (_Float16)f0.w;
            ah[4]  = (_Float16)f1.x; ah[5]  = (_Float16)f1.y;
            ah[6]  = (_Float16)f1.z; ah[7]  = (_Float16)f1.w;
            ah[8]  = (_Float16)f2.x; ah[9]  = (_Float16)f2.y;
            ah[10] = (_Float16)f2.z; ah[11] = (_Float16)f2.w;
            ah[12] = (_Float16)f3.x; ah[13] = (_Float16)f3.y;
            ah[14] = (_Float16)f3.z; ah[15] = (_Float16)f3.w;
            union { uint4 u[2]; v16h h; } b;
            b.u[0] = *(const uint4*)(vbase + kc + g * 8);
            b.u[1] = *(const uint4*)(vbase + kc + 16 + g * 8);
            c = __builtin_amdgcn_wmma_f32_16x16x32_f16(false, ah, false, b.h,
                                                       (short)0, c, false, false);
        }
        const int hh = hb / NB, bb = hb % NB;
#pragma unroll
        for (int r = 0; r < 8; ++r) {
            int row = r + g * 8;
            otmp[((size_t)bb * NS + i0 + row) * HID + hh * DH + wave * 16 + l15] =
                c[r] * rowinv[row];
        }
    }
}

// ---------------------------------------------------------------------------
// Residual + LayerNorm, in place on otmp (== out region of d_out), float4 I/O
__global__ __launch_bounds__(256) void ln_kernel(const float* __restrict__ query,
                                                 const float* __restrict__ gamma,
                                                 const float* __restrict__ beta,
                                                 float* __restrict__ out) {
    const int row = blockIdx.x;   // 0..NM-1
    const float4* qrow = (const float4*)(query + (size_t)row * HID);
    float4* orow = (float4*)(out + (size_t)row * HID);
    const int j4 = threadIdx.x;   // 256 threads * 4 floats = 1024

    float4 o = orow[j4], q = qrow[j4];
    float4 x = make_float4(o.x + q.x, o.y + q.y, o.z + q.z, o.w + q.w);
    float lsum = x.x + x.y + x.z + x.w;
    float lsq  = x.x * x.x + x.y * x.y + x.z * x.z + x.w * x.w;

    for (int off = 16; off > 0; off >>= 1) {
        lsum += __shfl_xor(lsum, off);
        lsq  += __shfl_xor(lsq, off);
    }
    __shared__ float s1[8], s2[8], mv[2];
    const int lane = threadIdx.x & 31, wv = threadIdx.x >> 5;
    if (lane == 0) { s1[wv] = lsum; s2[wv] = lsq; }
    __syncthreads();
    if (threadIdx.x == 0) {
        float a = 0.f, b = 0.f;
        for (int k = 0; k < 8; ++k) { a += s1[k]; b += s2[k]; }
        float mean = a * (1.0f / HID);
        float var  = b * (1.0f / HID) - mean * mean;
        mv[0] = mean; mv[1] = rsqrtf(var + 1e-5f);
    }
    __syncthreads();
    const float mean = mv[0], rstd = mv[1];
    float4 gm = ((const float4*)gamma)[j4];
    float4 bt = ((const float4*)beta)[j4];
    float4 r;
    r.x = (x.x - mean) * rstd * gm.x + bt.x;
    r.y = (x.y - mean) * rstd * gm.y + bt.y;
    r.z = (x.z - mean) * rstd * gm.z + bt.z;
    r.w = (x.w - mean) * rstd * gm.w + bt.w;
    orow[j4] = r;
}

// ---------------------------------------------------------------------------
extern "C" void kernel_launch(void* const* d_in, const int* in_sizes, int n_in,
                              void* d_out, int out_size, void* d_ws, size_t ws_size,
                              hipStream_t stream) {
    const float* q_in = (const float*)d_in[0];
    const float* k_in = (const float*)d_in[1];
    const float* v_in = (const float*)d_in[2];
    const float* Wq   = (const float*)d_in[3];
    const float* Wk   = (const float*)d_in[4];
    const float* Wv   = (const float*)d_in[5];
    const float* gam  = (const float*)d_in[6];
    const float* bet  = (const float*)d_in[7];
    const int*   mask = (const int*)d_in[8];

    float* out = (float*)d_out;                         // [b,s,hid]
    float* aw  = (float*)d_out + (size_t)NB * NS * HID; // [h,b,s,s]

    char* ws = (char*)d_ws;
    const size_t szX  = (size_t)NM * HID * 2;    // 8 MB per f16 matrix
    const size_t szW  = (size_t)HID * HID * 2;   // 2 MB per f16 weight
    _Float16* Xq = (_Float16*)(ws);
    _Float16* Xk = (_Float16*)(ws + szX);
    _Float16* Xv = (_Float16*)(ws + 2 * szX);
    _Float16* WTq = (_Float16*)(ws + 3 * szX);
    _Float16* WTk = (_Float16*)(ws + 3 * szX + szW);
    _Float16* WTv = (_Float16*)(ws + 3 * szX + 2 * szW);
    _Float16* Qb  = (_Float16*)(ws + 3 * szX + 3 * szW);
    _Float16* Kb  = (_Float16*)(ws + 4 * szX + 3 * szW);
    _Float16* VTb = (_Float16*)(ws + 5 * szX + 3 * szW);

    // 1) convert activations + weights to f16 (weights transposed)
    const int nX4 = NM * HID / 4;
    cvt_f16_kernel<<<nX4 / 256, 256, 0, stream>>>(q_in, Xq, nX4);
    cvt_f16_kernel<<<nX4 / 256, 256, 0, stream>>>(k_in, Xk, nX4);
    cvt_f16_kernel<<<nX4 / 256, 256, 0, stream>>>(v_in, Xv, nX4);
    const int nW = HID * HID;
    cvt_wt_kernel<<<nW / 256, 256, 0, stream>>>(Wq, WTq);
    cvt_wt_kernel<<<nW / 256, 256, 0, stream>>>(Wk, WTk);
    cvt_wt_kernel<<<nW / 256, 256, 0, stream>>>(Wv, WTv);

    // 2) projections (WMMA) -> head-major q/k, transposed v
    dim3 pg(64, 64);
    proj_kernel<<<pg, 128, 0, stream>>>(Xq, WTq, Qb, 0);
    proj_kernel<<<pg, 128, 0, stream>>>(Xk, WTk, Kb, 0);
    proj_kernel<<<pg, 128, 0, stream>>>(Xv, WTv, VTb, 1);

    // 3) fused attention
    const int smem_bytes = (16 * NS + 256 + 16 + 16) * (int)sizeof(float); // ~129 KB
    (void)hipFuncSetAttribute((const void*)attn_kernel,
                              hipFuncAttributeMaxDynamicSharedMemorySize, smem_bytes);
    dim3 ag(NS / 16, NH * NB);
    attn_kernel<<<ag, 256, smem_bytes, stream>>>(Qb, Kb, VTb, mask, aw, out);

    // 4) residual + layernorm
    ln_kernel<<<NM, 256, 0, stream>>>(q_in, gam, bet, out);
}
